// TreeLSTMCell_25254407701042
// MI455X (gfx1250) — compile-verified, matrix operand order checked by reading, not run on hardware
//
#include <hip/hip_runtime.h>
#include <hip/hip_bf16.h>

// ---------------------------------------------------------------------------
// TreeLSTM cell for MI455X (gfx1250, wave32, WMMA).
// Bandwidth-bound problem: scatter-add via global_atomic_add_f32, GEMMs via
// v_wmma_f32_16x16x32_bf16 (fp32 accumulate), fused in-register LSTM epilogue.
// ---------------------------------------------------------------------------

typedef __attribute__((ext_vector_type(16))) __bf16 v16bf;
typedef __attribute__((ext_vector_type(8)))  __bf16 v8bf;
typedef __attribute__((ext_vector_type(8)))  float  v8f;

#define HSZ      256     // hidden/X size
#define TILE_M   16
#define LDS_STR  264     // 256 + 8 bf16 pad -> 528B row stride, conflict-free b128 LDS loads

// --- helpers ---------------------------------------------------------------

__device__ __forceinline__ v16bf make_frag(const __bf16* p) {
  // 16 bf16 for one WMMA operand register group: 8 at p, 8 at p+16 elements.
  v8bf lo = *(const v8bf*)(p);
  v8bf hi = *(const v8bf*)(p + 16);
  return __builtin_shufflevector(lo, hi, 0,1,2,3,4,5,6,7,8,9,10,11,12,13,14,15);
}

__device__ __forceinline__ v8f wmma_bf16(v16bf a, v16bf b, v8f c) {
  // (neg_a, A, neg_b, B, c_mod, C, reuse_a, reuse_b)
  return __builtin_amdgcn_wmma_f32_16x16x32_bf16(false, a, false, b, (short)0, c,
                                                 false, false);
}

__device__ __forceinline__ float sigmoidf_(float x) {
  return 1.0f / (1.0f + __expf(-x));
}

// --- K0: zero h_sum / c_sum workspace --------------------------------------

__global__ __launch_bounds__(256) void tlstm_zero_kernel(float4* p, long n4) {
  long i = (long)blockIdx.x * blockDim.x + threadIdx.x;
  long stride = (long)gridDim.x * blockDim.x;
  float4 z = {0.f, 0.f, 0.f, 0.f};
  for (; i < n4; i += stride) p[i] = z;
}

// --- K1: fp32 weights -> bf16 packed weights -------------------------------
// Wx[768][256]  = W_iou
// Wh[1024][256] = [U_iou (768 rows); U_f_w (256 rows)]

__global__ __launch_bounds__(256) void tlstm_cvt_kernel(
    const float* __restrict__ Wiou, const float* __restrict__ Uiou,
    const float* __restrict__ Ufw, __bf16* __restrict__ Wx,
    __bf16* __restrict__ Wh) {
  int i = blockIdx.x * blockDim.x + threadIdx.x;   // 0 .. 1024*256-1
  const int NX = 768 * HSZ;
  if (i < NX) Wx[i] = (__bf16)Wiou[i];
  if (i < 1024 * HSZ) {
    float v = (i < NX) ? Uiou[i] : Ufw[i - NX];
    Wh[i] = (__bf16)v;
  }
}

// --- K2: edge scatter-add (segment_sum) ------------------------------------

__global__ __launch_bounds__(256) void tlstm_scatter_kernel(
    const float* __restrict__ h, const float* __restrict__ c,
    const int* __restrict__ src, const int* __restrict__ dst,
    float* __restrict__ hsum, float* __restrict__ csum) {
  int e = blockIdx.x;
  int t = threadIdx.x;                 // 0..255 = one feature per thread
  long s = (long)src[e] * HSZ;
  long d = (long)dst[e] * HSZ;
  atomicAdd(&hsum[d + t], h[s + t]);   // global_atomic_add_f32
  atomicAdd(&csum[d + t], c[s + t]);
}

// --- K3: fused WMMA GEMMs + LSTM epilogue ----------------------------------
// One block = 16 node rows, 8 waves. Wave w, set s handles output columns
// j0 = w*32 + s*16 .. j0+15 and accumulates the four 16x16 tiles {i,o,u,f}
// for those columns so the epilogue needs no cross-wave exchange.

__global__ __launch_bounds__(256) void tlstm_gemm_kernel(
    const float* __restrict__ x, const float* __restrict__ hsum,
    const float* __restrict__ csum,
    const __bf16* __restrict__ Wx, const __bf16* __restrict__ Wh,
    const float* __restrict__ biou, const float* __restrict__ ufb,
    float* __restrict__ hnew, float* __restrict__ cnew) {
  __shared__ __bf16 xs[TILE_M * LDS_STR];
  __shared__ __bf16 hsh[TILE_M * LDS_STR];

  const int t = threadIdx.x;
  const long rowBase = (long)blockIdx.x * TILE_M;

  // Stage x / h_sum tile (16x256 fp32 -> bf16 LDS)
#pragma unroll
  for (int j = 0; j < 16; ++j) {
    int idx = t + 256 * j;                  // 0..4095
    int r = idx >> 8;
    int col = idx & 255;
    long g = (rowBase + r) * HSZ + col;
    xs[r * LDS_STR + col]  = (__bf16)x[g];
    hsh[r * LDS_STR + col] = (__bf16)hsum[g];
  }
  __syncthreads();

  const int wave = t >> 5;
  const int lane = t & 31;
  const int half = lane >> 4;   // 0: K 0-7,16-23  1: K 8-15,24-31
  const int lr   = lane & 15;   // A: row; B: output column

#pragma unroll
  for (int set = 0; set < 2; ++set) {
    const int j0 = wave * 32 + set * 16;    // column base within H (0..255 step 16)
    v8f acc_i = {}, acc_o = {}, acc_u = {}, acc_f = {};

#pragma unroll
    for (int ks = 0; ks < 8; ++ks) {
      const int k0 = ks * 32;
      // A fragments from LDS (shared across the 7 WMMAs of this K-step)
      v16bf Ax = make_frag(&xs[lr * LDS_STR + k0 + half * 8]);
      v16bf Ah = make_frag(&hsh[lr * LDS_STR + k0 + half * 8]);
      const int koff = k0 + half * 8;
      // i-gate: x@W_iou[0:256]^T + h_sum@U_iou[0:256]^T
      acc_i = wmma_bf16(Ax, make_frag(&Wx[(0   + j0 + lr) * HSZ + koff]), acc_i);
      acc_i = wmma_bf16(Ah, make_frag(&Wh[(0   + j0 + lr) * HSZ + koff]), acc_i);
      // o-gate
      acc_o = wmma_bf16(Ax, make_frag(&Wx[(256 + j0 + lr) * HSZ + koff]), acc_o);
      acc_o = wmma_bf16(Ah, make_frag(&Wh[(256 + j0 + lr) * HSZ + koff]), acc_o);
      // u (cell candidate)
      acc_u = wmma_bf16(Ax, make_frag(&Wx[(512 + j0 + lr) * HSZ + koff]), acc_u);
      acc_u = wmma_bf16(Ah, make_frag(&Wh[(512 + j0 + lr) * HSZ + koff]), acc_u);
      // f-gate: h_sum @ U_f_w^T  (rows 768.. of Wh)
      acc_f = wmma_bf16(Ah, make_frag(&Wh[(768 + j0 + lr) * HSZ + koff]), acc_f);
    }

    // Epilogue: C/D layout => element r of acc holds (M = r + 8*half, Ncol = lr)
    const int col = j0 + lr;
    const float bi = biou[col];
    const float bo = biou[256 + col];
    const float bu = biou[512 + col];
    const float bf = ufb[col];

#pragma unroll
    for (int r = 0; r < 8; ++r) {
      long node = rowBase + r + half * 8;
      float cs = csum[node * HSZ + col];
      float iv = sigmoidf_(acc_i[r] + bi);
      float ov = sigmoidf_(acc_o[r] + bo);
      float uv = tanhf(acc_u[r] + bu);
      float fv = sigmoidf_(acc_f[r] + bf);
      float cn = iv * uv + fv * cs;
      float hn = ov * tanhf(cn);
      hnew[node * HSZ + col] = hn;
      cnew[node * HSZ + col] = cn;
    }
  }
}

// --- launcher --------------------------------------------------------------

extern "C" void kernel_launch(void* const* d_in, const int* in_sizes, int n_in,
                              void* d_out, int out_size, void* d_ws, size_t ws_size,
                              hipStream_t stream) {
  const float* x    = (const float*)d_in[0];
  const float* h    = (const float*)d_in[1];
  const float* c    = (const float*)d_in[2];
  const int*   src  = (const int*)d_in[3];
  const int*   dst  = (const int*)d_in[4];
  const float* Wiou = (const float*)d_in[5];
  const float* Uiou = (const float*)d_in[6];
  const float* biou = (const float*)d_in[7];
  const float* Ufw  = (const float*)d_in[8];
  const float* Ufb  = (const float*)d_in[9];

  const long N = in_sizes[0] / HSZ;       // 200000
  const int  E = in_sizes[3];             // 200000

  // Workspace layout: [h_sum f32 | c_sum f32 | Wx bf16 | Wh bf16]
  char* ws = (char*)d_ws;
  float*  hsum = (float*)ws;
  float*  csum = (float*)(ws + (size_t)N * HSZ * sizeof(float));
  __bf16* Wx   = (__bf16*)(ws + 2 * (size_t)N * HSZ * sizeof(float));
  __bf16* Wh   = Wx + (size_t)768 * HSZ;

  float* hnew = (float*)d_out;
  float* cnew = hnew + (size_t)N * HSZ;

  // K0: zero segment-sum buffers (2*N*256 floats)
  long n4 = (2L * N * HSZ) / 4;
  tlstm_zero_kernel<<<40960, 256, 0, stream>>>((float4*)hsum, n4);

  // K1: weights -> bf16 (1024*256 threads covers both packed matrices)
  tlstm_cvt_kernel<<<(1024 * HSZ) / 256, 256, 0, stream>>>(Wiou, Uiou, Ufw, Wx, Wh);

  // K2: edge scatter-add (one block per edge, one feature per thread)
  tlstm_scatter_kernel<<<E, 256, 0, stream>>>(h, c, src, dst, hsum, csum);

  // K3: fused GEMM + LSTM epilogue (N is an exact multiple of 16)
  tlstm_gemm_kernel<<<(int)(N / TILE_M), 256, 0, stream>>>(
      x, hsum, csum, Wx, Wh, biou, Ufb, hnew, cnew);
}
